// MultiHeadAttentionLayer_15590731284509
// MI455X (gfx1250) — matrix operand order repaired and checked by reading, not compiled, not run on hardware
//
#include <hip/hip_runtime.h>
#include <hip/hip_bf16.h>

// Problem constants (from reference)
#define B_   2
#define L_   2048
#define H_   1024
#define NH_  16
#define HD_  64
#define ML_  (B_ * L_)          // 4096 total rows for the projections
#define LOG2E 1.4426950408889634f

typedef __attribute__((ext_vector_type(16))) _Float16 v16h;
typedef __attribute__((ext_vector_type(8)))  _Float16 v8h;
typedef __attribute__((ext_vector_type(8)))  float    v8f;
typedef __attribute__((ext_vector_type(4)))  float    v4f;
typedef unsigned int u32x4 __attribute__((ext_vector_type(4)));
typedef int          i32x4 __attribute__((ext_vector_type(4)));
typedef int          i32x8 __attribute__((ext_vector_type(8)));

#if __has_include(<hip/amd_detail/amd_gfx1250_TDM.h>)
#define TDM_6ARG 1
#endif

union F16x16 { v16h v; v8h h[2]; };

__device__ __forceinline__ v8f zero8() {
    v8f r = {0.f, 0.f, 0.f, 0.f, 0.f, 0.f, 0.f, 0.f};
    return r;
}

__device__ __forceinline__ v8f wmma_f16(v16h a, v16h b, v8f c) {
    // D = A(16x32 f16) * B(32x16 f16) + C(16x16 f32)
    return __builtin_amdgcn_wmma_f32_16x16x32_f16(
        false, a, false, b, (short)0, c, false, false);
}

__device__ __forceinline__ v8h cvt8(const float* p) {
    v4f a = *(const v4f*)p;
    v4f b = *(const v4f*)(p + 4);
    v8h r;
    r[0] = (_Float16)a[0]; r[1] = (_Float16)a[1];
    r[2] = (_Float16)a[2]; r[3] = (_Float16)a[3];
    r[4] = (_Float16)b[0]; r[5] = (_Float16)b[1];
    r[6] = (_Float16)b[2]; r[7] = (_Float16)b[3];
    return r;
}

// 16 contiguous halfs from LDS as two 16B-aligned v8h reads (ds_load_b128 x2)
__device__ __forceinline__ v16h ldsv16(const _Float16* p) {
    F16x16 t;
    t.h[0] = *(const v8h*)p;
    t.h[1] = *(const v8h*)(p + 8);
    return t.v;
}

// ---------------------------------------------------------------------------
// TDM descriptor helpers (CDNA5 ISA §8.3/§8.4). D# group1 bitfield packing.
// ---------------------------------------------------------------------------
__device__ __forceinline__ i32x8 tdm_group1(unsigned pad_int, unsigned pad_amt,
                                            unsigned long long tdim0,
                                            unsigned long long tdim1,
                                            unsigned tile0, unsigned tile1,
                                            unsigned long long stride0,
                                            unsigned long long stride1) {
    // [17:16]=data_size(1 -> 2B)  [20]=pad_enable  [24:22]=pad_interval
    // [31:25]=pad_amount  [79:48]=tensor_dim0  [111:80]=tensor_dim1
    // [127:112]=tile_dim0 [143:128]=tile_dim1  [207:160]=dim0_stride
    // [255:208]=dim1_stride
    unsigned dw0 = (1u << 16) | (1u << 20) | (pad_int << 22) | (pad_amt << 25);
    unsigned dw1 = (unsigned)((tdim0 & 0xFFFFull) << 16);
    unsigned dw2 = (unsigned)((tdim0 >> 16) & 0xFFFFull) |
                   (unsigned)((tdim1 & 0xFFFFull) << 16);
    unsigned dw3 = (unsigned)((tdim1 >> 16) & 0xFFFFull) | (tile0 << 16);
    unsigned dw4 = tile1;                      // tile_dim2 = 0
    unsigned dw5 = (unsigned)(stride0 & 0xFFFFFFFFull);
    unsigned dw6 = (unsigned)((stride0 >> 32) & 0xFFFFull) |
                   (unsigned)((stride1 & 0xFFFFull) << 16);
    unsigned dw7 = (unsigned)((stride1 >> 16) & 0xFFFFFFFFull);
    i32x8 g;
    g[0] = (int)dw0; g[1] = (int)dw1; g[2] = (int)dw2; g[3] = (int)dw3;
    g[4] = (int)dw4; g[5] = (int)dw5; g[6] = (int)dw6; g[7] = (int)dw7;
    return g;
}

__device__ __forceinline__ void tdm_load(unsigned lds_addr, const void* gaddr,
                                         i32x8 g1) {
    unsigned long long ga = (unsigned long long)(size_t)gaddr;
    u32x4 g0;
    g0[0] = 1u;                                   // count=1 valid descriptor
    g0[1] = lds_addr;                             // LDS byte offset
    g0[2] = (unsigned)(ga & 0xFFFFFFFFull);       // global_addr[31:0]
    g0[3] = (unsigned)((ga >> 32) & 0x01FFFFFFull) | (2u << 30); // [56:32]|type=2
    i32x4 z4 = {0, 0, 0, 0};
#ifdef TDM_6ARG
    i32x8 z8 = {0, 0, 0, 0, 0, 0, 0, 0};
    __builtin_amdgcn_tensor_load_to_lds(g0, g1, z4, z4, z8, 0);
#else
    __builtin_amdgcn_tensor_load_to_lds(g0, g1, z4, z4, 0);
#endif
}

// ---------------------------------------------------------------------------
// Kernel 1: convert the three HxH fp32 weight matrices to f16 (concatenated).
// ---------------------------------------------------------------------------
__global__ __launch_bounds__(256)
void cvt_weights(const float* __restrict__ wq, const float* __restrict__ wk,
                 const float* __restrict__ wv, _Float16* __restrict__ wh) {
    int i = blockIdx.x * 256 + threadIdx.x;
    const int n = H_ * H_;
    if (i < n)            wh[i] = (_Float16)wq[i];
    else if (i < 2 * n)   wh[i] = (_Float16)wk[i - n];
    else                  wh[i] = (_Float16)wv[i - 2 * n];
}

// ---------------------------------------------------------------------------
// Kernel 2: QKV projection GEMM  Y = X * W^T + b  via v_wmma_f32_16x16x32_f16.
// Block tile 128(M) x 64(N), 8 waves arranged 4x2, each wave 32x32 (2x2 frags).
// Q is pre-scaled by 1/sqrt(HD); V is written transposed (B,NH,HD,L).
// ---------------------------------------------------------------------------
__global__ __launch_bounds__(256)
void qkv_proj(const float* __restrict__ q_in, const float* __restrict__ k_in,
              const float* __restrict__ v_in, const _Float16* __restrict__ wh,
              const float* __restrict__ bq, const float* __restrict__ bk,
              const float* __restrict__ bv,
              _Float16* __restrict__ Qh, _Float16* __restrict__ Kh,
              _Float16* __restrict__ Vt) {
    const int z = blockIdx.z;   // 0=Q, 1=K, 2=V
    const float* X        = (z == 0) ? q_in : ((z == 1) ? k_in : v_in);
    const _Float16* W     = wh + (size_t)z * H_ * H_;
    const float* bias     = (z == 0) ? bq : ((z == 1) ? bk : bv);

    const int lane = threadIdx.x & 31;
    const int wv_  = threadIdx.x >> 5;          // wave id 0..7
    const int wm   = wv_ & 3, wn = wv_ >> 2;
    const int m_w  = blockIdx.x * 128 + wm * 32;
    const int n_w  = blockIdx.y * 64  + wn * 32;
    const int lrow = lane & 15, lhi = lane >> 4;

    v8f acc[2][2];
    acc[0][0] = zero8(); acc[0][1] = zero8();
    acc[1][0] = zero8(); acc[1][1] = zero8();

    for (int kk = 0; kk < H_; kk += 32) {
        F16x16 afr[2], bfr[2];
        #pragma unroll
        for (int mi = 0; mi < 2; ++mi) {
            const float* ap = X + (size_t)(m_w + mi * 16 + lrow) * H_ + kk + lhi * 16;
            afr[mi].h[0] = cvt8(ap);
            afr[mi].h[1] = cvt8(ap + 8);
        }
        #pragma unroll
        for (int ni = 0; ni < 2; ++ni) {
            const _Float16* bp = W + (size_t)(n_w + ni * 16 + lrow) * H_ + kk + lhi * 16;
            bfr[ni].v = *(const v16h*)bp;
        }
        #pragma unroll
        for (int mi = 0; mi < 2; ++mi)
            #pragma unroll
            for (int ni = 0; ni < 2; ++ni)
                acc[mi][ni] = wmma_f16(afr[mi].v, bfr[ni].v, acc[mi][ni]);
    }

    const float scale = (z == 0) ? 0.125f : 1.0f;   // fold 1/sqrt(64) into Q
    #pragma unroll
    for (int ni = 0; ni < 2; ++ni) {
        const int n    = n_w + ni * 16 + lrow;
        const float bb = bias[n];
        const int hh = n / HD_, dd = n % HD_;
        #pragma unroll
        for (int mi = 0; mi < 2; ++mi) {
            #pragma unroll
            for (int r = 0; r < 8; ++r) {
                const int m  = m_w + mi * 16 + r + lhi * 8;
                const int bi = m / L_, ll = m & (L_ - 1);
                const _Float16 val = (_Float16)((acc[mi][ni][r] + bb) * scale);
                if (z == 2) {
                    Vt[(((size_t)bi * NH_ + hh) * HD_ + dd) * L_ + ll] = val;
                } else {
                    _Float16* dst = (z == 0) ? Qh : Kh;
                    dst[(((size_t)bi * NH_ + hh) * L_ + ll) * HD_ + dd] = val;
                }
            }
        }
    }
}

// ---------------------------------------------------------------------------
// Kernel 3: flash attention. Workgroup = (b,h) x 128 query rows, 8 waves x 16
// rows. K/V tiles (32 keys) are staged to LDS by the Tensor Data Mover
// (double-buffered, TENSORcnt), with TDM padding to de-conflict LDS banks:
//   K tile rows: 64 halfs + 8 pad  -> stride 144B (banks 0,36,8,44,...)
//   V tile rows: 32 halfs + 8 pad  -> stride  80B (banks 0,20,40,60,...)
// Softmax P relayout C->A goes through a wave-private LDS tile.
// ---------------------------------------------------------------------------
__global__ __launch_bounds__(256)
void attn(const _Float16* __restrict__ Qh, const _Float16* __restrict__ Kh,
          const _Float16* __restrict__ Vt, const float* __restrict__ key_mask,
          float* __restrict__ out) {
    __shared__ _Float16 Ktile[2][2336];     // 32*72 + slack
    __shared__ _Float16 Vtile[2][2592];     // 64*40 + slack
    __shared__ _Float16 plds[8][16 * 40];   // per-wave 16x32 P tile

    const int lane = threadIdx.x & 31;
    const int wv_  = threadIdx.x >> 5;
    const int bh   = blockIdx.x;            // 0..B*NH-1
    const int b    = bh / NH_, hh = bh % NH_;
    const int lrow = lane & 15, lhi = lane >> 4;
    const int q0   = blockIdx.y * 128 + wv_ * 16;

    const _Float16* Qb = Qh + ((size_t)bh * L_ + q0) * HD_;
    const _Float16* Kb = Kh + (size_t)bh * L_ * HD_;
    const _Float16* Vb = Vt + (size_t)bh * HD_ * L_;
    const float*    mb = key_mask + (size_t)b * L_;

    // TDM descriptors (uniform). K: 1-D 2048 halfs, pad 4dw per 32dw.
    // V: 2-D 32x64 halfs, row stride L_, pad 4dw per 16dw.
    const i32x8 g1K = tdm_group1(4, 3, (unsigned long long)L_ * HD_, 1ull,
                                 32 * HD_, 1, (unsigned long long)L_ * HD_, 0ull);
    const i32x8 g1V = tdm_group1(3, 3, (unsigned long long)L_,
                                 (unsigned long long)HD_, 32, HD_,
                                 (unsigned long long)L_, 0ull);
    const unsigned ldsK[2] = { (unsigned)(size_t)&Ktile[0][0],
                               (unsigned)(size_t)&Ktile[1][0] };
    const unsigned ldsV[2] = { (unsigned)(size_t)&Vtile[0][0],
                               (unsigned)(size_t)&Vtile[1][0] };

    // Q A-fragments resident for the whole K loop (d-chunks 0..31, 32..63)
    F16x16 aq[2];
    #pragma unroll
    for (int c = 0; c < 2; ++c) {
        const _Float16* qp = Qb + (size_t)lrow * HD_ + c * 32 + lhi * 16;
        aq[c].h[0] = *(const v8h*)qp;
        aq[c].h[1] = *(const v8h*)(qp + 8);
    }

    v8f o[4];
    #pragma unroll
    for (int ni = 0; ni < 4; ++ni) o[ni] = zero8();
    float mrow[8], lsum[8];
    #pragma unroll
    for (int r = 0; r < 8; ++r) { mrow[r] = -1e30f; lsum[r] = 0.f; }

    const int NT = L_ / 32;                 // 64 key tiles
    if (wv_ == 0) {                         // prologue: stage tile 0
        tdm_load(ldsK[0], Kb, g1K);
        tdm_load(ldsV[0], Vb, g1V);
    }

    for (int it = 0; it < NT; ++it) {
        const int kk  = it * 32;
        const int buf = it & 1;

        if (wv_ == 0) {
            if (it + 1 < NT) {              // prefetch next tile via TDM
                tdm_load(ldsK[buf ^ 1], Kb + (size_t)(kk + 32) * HD_, g1K);
                tdm_load(ldsV[buf ^ 1], Vb + (kk + 32), g1V);
                __builtin_amdgcn_s_wait_tensorcnt((short)2);  // oldest K,V done
            } else {
                __builtin_amdgcn_s_wait_tensorcnt((short)0);
            }
        }
        __syncthreads();                    // tile `buf` visible to all waves

        // ---- scores: S = (Q/8) * K^T, two 16-key groups, from LDS ----
        const _Float16* kt = &Ktile[buf][0];
        v8f s0 = zero8(), s1 = zero8();
        {
            const _Float16* kp0 = kt + (lrow)      * 72 + lhi * 16;
            const _Float16* kp1 = kt + (16 + lrow) * 72 + lhi * 16;
            s0 = wmma_f16(aq[0].v, ldsv16(kp0),      s0);
            s0 = wmma_f16(aq[1].v, ldsv16(kp0 + 32), s0);
            s1 = wmma_f16(aq[0].v, ldsv16(kp1),      s1);
            s1 = wmma_f16(aq[1].v, ldsv16(kp1 + 32), s1);
        }
        const float madd0 = (1.0f - mb[kk + lrow])      * -10000.0f;
        const float madd1 = (1.0f - mb[kk + 16 + lrow]) * -10000.0f;

        // ---- online softmax (base-2); C-frag row = r + 8*(lane/16) ----
        float alpha[8];
        #pragma unroll
        for (int r = 0; r < 8; ++r) {
            float t0 = (s0[r] + madd0) * LOG2E;
            float t1 = (s1[r] + madd1) * LOG2E;
            float rm = fmaxf(t0, t1);
            rm = fmaxf(rm, __shfl_xor(rm, 1));
            rm = fmaxf(rm, __shfl_xor(rm, 2));
            rm = fmaxf(rm, __shfl_xor(rm, 4));
            rm = fmaxf(rm, __shfl_xor(rm, 8));
            float mnew = fmaxf(mrow[r], rm);
            float a_   = exp2f(mrow[r] - mnew);
            float p0   = exp2f(t0 - mnew);
            float p1   = exp2f(t1 - mnew);
            float ps   = p0 + p1;
            ps += __shfl_xor(ps, 1);
            ps += __shfl_xor(ps, 2);
            ps += __shfl_xor(ps, 4);
            ps += __shfl_xor(ps, 8);
            lsum[r] = lsum[r] * a_ + ps;
            mrow[r] = mnew;
            alpha[r] = a_;
            const int prow = r + lhi * 8;
            plds[wv_][prow * 40 + lrow]      = (_Float16)p0;
            plds[wv_][prow * 40 + 16 + lrow] = (_Float16)p1;
        }
        asm volatile("s_wait_dscnt 0" ::: "memory");  // wave-private LDS RAW

        F16x16 ap;                       // P as A-fragment
        ap.h[0] = *(const v8h*)&plds[wv_][lrow * 40 + lhi * 16];
        ap.h[1] = *(const v8h*)&plds[wv_][lrow * 40 + lhi * 16 + 8];

        // ---- ctx = ctx*alpha + P * V, B-frags from padded LDS V tile ----
        const _Float16* vt = &Vtile[buf][0];
        #pragma unroll
        for (int ni = 0; ni < 4; ++ni) {
            #pragma unroll
            for (int r = 0; r < 8; ++r) o[ni][r] *= alpha[r];
            v16h bv16 = ldsv16(vt + (size_t)(ni * 16 + lrow) * 40 + lhi * 16);
            o[ni] = wmma_f16(ap.v, bv16, o[ni]);
        }
        __syncthreads();                 // done reading buf before its reuse
    }

    // ---- epilogue: normalize and store fp32 (B, L, H) ----
    float inv[8];
    #pragma unroll
    for (int r = 0; r < 8; ++r) inv[r] = (lsum[r] > 0.f) ? 1.0f / lsum[r] : 0.f;
    #pragma unroll
    for (int ni = 0; ni < 4; ++ni) {
        const int dd = ni * 16 + lrow;
        #pragma unroll
        for (int r = 0; r < 8; ++r) {
            const int qr = q0 + r + lhi * 8;
            out[((size_t)b * L_ + qr) * H_ + hh * HD_ + dd] = o[ni][r] * inv[r];
        }
    }
}

// ---------------------------------------------------------------------------
extern "C" void kernel_launch(void* const* d_in, const int* in_sizes, int n_in,
                              void* d_out, int out_size, void* d_ws, size_t ws_size,
                              hipStream_t stream) {
    (void)in_sizes; (void)n_in; (void)out_size; (void)ws_size;
    const float* q_in = (const float*)d_in[0];
    const float* k_in = (const float*)d_in[1];
    const float* v_in = (const float*)d_in[2];
    const float* mask = (const float*)d_in[3];
    const float* wq   = (const float*)d_in[4];
    const float* bq   = (const float*)d_in[5];
    const float* wk   = (const float*)d_in[6];
    const float* bk   = (const float*)d_in[7];
    const float* wv   = (const float*)d_in[8];
    const float* bv   = (const float*)d_in[9];
    float* out = (float*)d_out;

    // Workspace layout (f16): Q 8MB | K 8MB | V^T 8MB | weights 6MB = 30MB
    char* ws = (char*)d_ws;
    _Float16* Qh = (_Float16*)(ws);
    _Float16* Kh = (_Float16*)(ws + ((size_t)8  << 20));
    _Float16* Vt = (_Float16*)(ws + ((size_t)16 << 20));
    _Float16* Wh = (_Float16*)(ws + ((size_t)24 << 20));

    cvt_weights<<<dim3(3 * H_ * H_ / 256), dim3(256), 0, stream>>>(wq, wk, wv, Wh);
    qkv_proj<<<dim3(ML_ / 128, H_ / 64, 3), dim3(256), 0, stream>>>(
        q_in, k_in, v_in, Wh, bq, bk, bv, Qh, Kh, Vt);
    attn<<<dim3(B_ * NH_, L_ / 128), dim3(256), 0, stream>>>(Qh, Kh, Vt, mask, out);
}
